// AngleBlock_1245540516166
// MI455X (gfx1250) — matrix operand order repaired and checked by reading, not compiled, not run on hardware
//
#include <hip/hip_runtime.h>

#define EDGE_DIM 64
#define HIDDEN   128
#define IN_DIM   132
#define KPAD     160   // 132 padded to 5 K-steps of 32
#define LN_EPS   1e-5f

typedef _Float16 half4  __attribute__((ext_vector_type(4)));
typedef _Float16 half8  __attribute__((ext_vector_type(8)));
typedef _Float16 v16h   __attribute__((ext_vector_type(16)));
typedef float    v8f    __attribute__((ext_vector_type(8)));

// -------- LDS layout (dynamic, 132 KB total; gfx1250 WGP has 320 KB) --------
constexpr int SM_W1T   = 0;                       // 128 x 160 f16 (W1^T, K-padded)
constexpr int SM_W2T   = SM_W1T + 128*160*2;      // 128 x 128 f16
constexpr int SM_W3T   = SM_W2T + 128*128*2;      // 64  x 128 f16
constexpr int SM_X     = SM_W3T + 64*128*2;       // 8 waves x [16 x 160] f16 stage
constexpr int SM_VEC   = SM_X   + 8*16*160*2;     // b1,g1,be1,b2,g2,be2 : 6x128 f32
constexpr int SM_IDX   = SM_VEC + 6*128*4;        // e_ij per wave-row
constexpr int SM_KJ    = SM_IDX + 8*16*4;         // e_kj per wave-row
constexpr int SM_TOTAL = SM_KJ  + 8*16*4;         // = 135168 B

// A-fragment (16x32 f16): lane<16 row=lane K={k*32+0..7, +16..23}; lane>=16 K base +8
__device__ __forceinline__ v16h frag_load_A(const _Float16* __restrict__ base,
                                            int row, int stride, int kstep, int halfsel) {
  const _Float16* p = base + row*stride + kstep*32 + halfsel*8;
  half8 lo = *(const half8*)(p);
  half8 hi = *(const half8*)(p + 16);
  return __builtin_shufflevector(lo, hi, 0,1,2,3,4,5,6,7,8,9,10,11,12,13,14,15);
}

// B-fragment (32x16 f16) from W^T[row=n][K]: lane<16 K={k*32+0..7, +16..23}; lane>=16 K base +8
__device__ __forceinline__ v16h frag_load_B(const _Float16* __restrict__ base,
                                            int row, int stride, int kstep, int halfsel) {
  const _Float16* p = base + row*stride + kstep*32 + halfsel*8;
  half8 lo = *(const half8*)(p);
  half8 hi = *(const half8*)(p + 16);
  return __builtin_shufflevector(lo, hi, 0,1,2,3,4,5,6,7,8,9,10,11,12,13,14,15);
}

__device__ __forceinline__ float silu_f(float x) {
  return x / (1.0f + __expf(-x));
}

__global__ void __launch_bounds__(256)
triplet_mlp_kernel(const float* __restrict__ edge_feat,
                   const int*   __restrict__ tidx,
                   const float* __restrict__ geo,
                   const float* __restrict__ W1, const float* __restrict__ b1,
                   const float* __restrict__ g1, const float* __restrict__ be1,
                   const float* __restrict__ W2, const float* __restrict__ b2,
                   const float* __restrict__ g2, const float* __restrict__ be2,
                   const float* __restrict__ W3,
                   float* __restrict__ acc, float* __restrict__ cnt,
                   int T, int numTiles) {
  extern __shared__ char smem[];
  _Float16* sW1t = (_Float16*)(smem + SM_W1T);
  _Float16* sW2t = (_Float16*)(smem + SM_W2T);
  _Float16* sW3t = (_Float16*)(smem + SM_W3T);
  _Float16* sX   = (_Float16*)(smem + SM_X);
  float*    sB1  = (float*)(smem + SM_VEC);
  float*    sG1  = sB1 + 128;
  float*    sBe1 = sB1 + 256;
  float*    sB2  = sB1 + 384;
  float*    sG2  = sB1 + 512;
  float*    sBe2 = sB1 + 640;
  int*      sIdx = (int*)(smem + SM_IDX);
  int*      sKj  = (int*)(smem + SM_KJ);

  const int tid     = threadIdx.x;
  const int wv      = tid >> 5;
  const int lane    = tid & 31;
  const int r16     = lane & 15;
  const int halfsel = lane >> 4;

  // ---- one-time per block: transpose weights to f16 in LDS (coalesced reads) ----
  for (int i = tid; i < 128*160; i += 256) {          // W1^T [n][k], K padded to 160
    int n = i & 127, k = i >> 7;
    float v = (k < IN_DIM) ? W1[k*HIDDEN + n] : 0.0f;
    sW1t[n*KPAD + k] = (_Float16)v;
  }
  for (int i = tid; i < 128*128; i += 256) {          // W2^T [n][k]
    int n = i & 127, k = i >> 7;
    sW2t[n*HIDDEN + k] = (_Float16)W2[k*HIDDEN + n];
  }
  for (int i = tid; i < 64*128; i += 256) {           // W3^T [n][k]
    int n = i & 63, k = i >> 6;
    sW3t[n*HIDDEN + k] = (_Float16)W3[k*EDGE_DIM + n];
  }
  for (int i = tid; i < 128; i += 256) {
    sB1[i] = b1[i]; sG1[i] = g1[i]; sBe1[i] = be1[i];
    sB2[i] = b2[i]; sG2[i] = g2[i]; sBe2[i] = be2[i];
  }
  // zero the K-pad region of every wave's stage (written once, never dirtied)
  for (int i = tid; i < 8*16*28; i += 256) {
    int w = i / 448, rem = i % 448;
    int r = rem / 28, k = IN_DIM + rem % 28;
    sX[w*16*KPAD + r*KPAD + k] = (_Float16)0.0f;
  }
  __syncthreads();

  _Float16* xstage = sX + wv * (16*KPAD);

  for (int tile = blockIdx.x; tile < numTiles; tile += gridDim.x) {
    const int t0 = tile*128 + wv*16;   // this wave's 16 triplets

    // ---- gather indices + counts ----
    if (lane < 16) {
      int t = t0 + lane; int tc = (t < T) ? t : (T - 1);
      int eij = tidx[2*tc];
      sIdx[wv*16 + lane] = eij;
      sKj [wv*16 + lane] = tidx[2*tc + 1];
      if (t < T) atomicAdd(cnt + eij, 1.0f);
    }

    // ---- gather x rows (f32 -> f16 stage), float4-vectorized ----
    for (int r = 0; r < 16; ++r) {
      int eij = sIdx[wv*16 + r];
      int ekj = sKj [wv*16 + r];
      int t = t0 + r; int tc = (t < T) ? t : (T - 1);
      const float4* fij = (const float4*)(edge_feat + (size_t)eij*EDGE_DIM);
      const float4* fkj = (const float4*)(edge_feat + (size_t)ekj*EDGE_DIM);
      const float4* g4  = (const float4*)(geo + (size_t)tc*4);
      for (int c = lane; c < 33; c += 32) {
        float4 v = (c < 16) ? fij[c] : (c < 32) ? fkj[c-16] : g4[0];
        half4 h = { (_Float16)v.x, (_Float16)v.y, (_Float16)v.z, (_Float16)v.w };
        *(half4*)(xstage + r*KPAD + c*4) = h;
      }
    }

    // ================= layer 1: x(16x132) @ W1 -> SiLU -> LN =================
    {
      v16h av[5];
      #pragma unroll
      for (int k = 0; k < 5; ++k) av[k] = frag_load_A(xstage, r16, KPAD, k, halfsel);
      v8f cc[8];
      #pragma unroll
      for (int n = 0; n < 8; ++n) {
        int col = n*16 + r16;
        float bb = sB1[col];
        v8f c = { bb, bb, bb, bb, bb, bb, bb, bb };
        #pragma unroll
        for (int k = 0; k < 5; ++k) {
          v16h bv = frag_load_B(sW1t, col, KPAD, k, halfsel);
          c = __builtin_amdgcn_wmma_f32_16x16x32_f16(false, av[k], false, bv,
                                                     (short)0, c, false, false);
        }
        cc[n] = c;
      }
      #pragma unroll
      for (int n = 0; n < 8; ++n)
        #pragma unroll
        for (int j = 0; j < 8; ++j) cc[n][j] = silu_f(cc[n][j]);
      #pragma unroll
      for (int j = 0; j < 8; ++j) {          // LN per row (row = j + halfsel*8)
        float s = 0.0f, q = 0.0f;
        #pragma unroll
        for (int n = 0; n < 8; ++n) { float v = cc[n][j]; s += v; q += v*v; }
        #pragma unroll
        for (int off = 1; off < 16; off <<= 1) {
          s += __shfl_xor(s, off, 32); q += __shfl_xor(q, off, 32);
        }
        float mu = s * (1.0f/128.0f);
        float rs = rsqrtf(q * (1.0f/128.0f) - mu*mu + LN_EPS);
        int row = j + halfsel*8;
        #pragma unroll
        for (int n = 0; n < 8; ++n) {
          int col = n*16 + r16;
          xstage[row*KPAD + col] = (_Float16)((cc[n][j] - mu) * rs * sG1[col] + sBe1[col]);
        }
      }
    }

    // ================= layer 2: h(16x128) @ W2 -> SiLU -> LN =================
    {
      v16h av[4];
      #pragma unroll
      for (int k = 0; k < 4; ++k) av[k] = frag_load_A(xstage, r16, KPAD, k, halfsel);
      v8f cc[8];
      #pragma unroll
      for (int n = 0; n < 8; ++n) {
        int col = n*16 + r16;
        float bb = sB2[col];
        v8f c = { bb, bb, bb, bb, bb, bb, bb, bb };
        #pragma unroll
        for (int k = 0; k < 4; ++k) {
          v16h bv = frag_load_B(sW2t, col, HIDDEN, k, halfsel);
          c = __builtin_amdgcn_wmma_f32_16x16x32_f16(false, av[k], false, bv,
                                                     (short)0, c, false, false);
        }
        cc[n] = c;
      }
      #pragma unroll
      for (int n = 0; n < 8; ++n)
        #pragma unroll
        for (int j = 0; j < 8; ++j) cc[n][j] = silu_f(cc[n][j]);
      #pragma unroll
      for (int j = 0; j < 8; ++j) {
        float s = 0.0f, q = 0.0f;
        #pragma unroll
        for (int n = 0; n < 8; ++n) { float v = cc[n][j]; s += v; q += v*v; }
        #pragma unroll
        for (int off = 1; off < 16; off <<= 1) {
          s += __shfl_xor(s, off, 32); q += __shfl_xor(q, off, 32);
        }
        float mu = s * (1.0f/128.0f);
        float rs = rsqrtf(q * (1.0f/128.0f) - mu*mu + LN_EPS);
        int row = j + halfsel*8;
        #pragma unroll
        for (int n = 0; n < 8; ++n) {
          int col = n*16 + r16;
          xstage[row*KPAD + col] = (_Float16)((cc[n][j] - mu) * rs * sG2[col] + sBe2[col]);
        }
      }
    }

    // ========== layer 3: m(16x128) @ W3 -> f32 atomic scatter-add ==========
    {
      v16h av[4];
      #pragma unroll
      for (int k = 0; k < 4; ++k) av[k] = frag_load_A(xstage, r16, KPAD, k, halfsel);
      v8f cc[4];
      #pragma unroll
      for (int n = 0; n < 4; ++n) {
        v8f c = { 0.f, 0.f, 0.f, 0.f, 0.f, 0.f, 0.f, 0.f };
        #pragma unroll
        for (int k = 0; k < 4; ++k) {
          v16h bv = frag_load_B(sW3t, n*16 + r16, HIDDEN, k, halfsel);
          c = __builtin_amdgcn_wmma_f32_16x16x32_f16(false, av[k], false, bv,
                                                     (short)0, c, false, false);
        }
        cc[n] = c;
      }
      #pragma unroll
      for (int j = 0; j < 8; ++j) {
        int row = j + halfsel*8;
        if (t0 + row < T) {
          int e = sIdx[wv*16 + row];
          size_t base = (size_t)e * EDGE_DIM + r16;
          #pragma unroll
          for (int n = 0; n < 4; ++n) atomicAdd(acc + base + n*16, cc[n][j]);
        }
      }
    }
  }
}

// out = LN(edge_feat + acc/max(cnt,1) + b3);  one wave per edge, 2 cols/lane
__global__ void __launch_bounds__(256)
edge_finalize_kernel(const float* __restrict__ edge_feat,
                     const float* __restrict__ acc,
                     const float* __restrict__ cnt,
                     const float* __restrict__ b3,
                     const float* __restrict__ gn,
                     const float* __restrict__ bn,
                     float* __restrict__ out, int E) {
  int wv = threadIdx.x >> 5;
  int lane = threadIdx.x & 31;
  int e = blockIdx.x * 8 + wv;
  if (e >= E) return;
  float inv = 1.0f / fmaxf(cnt[e], 1.0f);
  int c0 = lane * 2;
  size_t base = (size_t)e * EDGE_DIM + c0;
  float2 f = *(const float2*)(edge_feat + base);
  float2 a = *(const float2*)(acc + base);
  float v0 = f.x + a.x * inv + b3[c0];
  float v1 = f.y + a.y * inv + b3[c0 + 1];
  float s = v0 + v1, q = v0*v0 + v1*v1;
  #pragma unroll
  for (int off = 1; off < 32; off <<= 1) {
    s += __shfl_xor(s, off, 32); q += __shfl_xor(q, off, 32);
  }
  float mu = s * (1.0f/64.0f);
  float rs = rsqrtf(q * (1.0f/64.0f) - mu*mu + LN_EPS);
  float2 o;
  o.x = (v0 - mu) * rs * gn[c0] + bn[c0];
  o.y = (v1 - mu) * rs * gn[c0 + 1] + bn[c0 + 1];
  *(float2*)(out + base) = o;
}

extern "C" void kernel_launch(void* const* d_in, const int* in_sizes, int n_in,
                              void* d_out, int out_size, void* d_ws, size_t ws_size,
                              hipStream_t stream) {
  const float* edge_feat = (const float*)d_in[0];
  const int*   tidx      = (const int*)  d_in[1];
  const float* geo       = (const float*)d_in[2];
  const float* W1  = (const float*)d_in[3];
  const float* b1  = (const float*)d_in[4];
  const float* g1  = (const float*)d_in[5];
  const float* be1 = (const float*)d_in[6];
  const float* W2  = (const float*)d_in[7];
  const float* b2  = (const float*)d_in[8];
  const float* g2  = (const float*)d_in[9];
  const float* be2 = (const float*)d_in[10];
  const float* W3  = (const float*)d_in[11];
  const float* b3  = (const float*)d_in[12];
  const float* gn  = (const float*)d_in[13];
  const float* bn  = (const float*)d_in[14];

  const int E = in_sizes[0] / EDGE_DIM;
  const int T = in_sizes[1] / 2;

  float* acc = (float*)d_ws;                 // E x 64 f32 accumulator
  float* cnt = acc + (size_t)E * EDGE_DIM;   // E   f32 counts
  hipMemsetAsync(d_ws, 0, ((size_t)E * EDGE_DIM + (size_t)E) * sizeof(float), stream);

  const int numTiles = (T + 127) / 128;
  int blocks = numTiles < 2048 ? numTiles : 2048;
  hipLaunchKernelGGL(triplet_mlp_kernel, dim3(blocks), dim3(256), SM_TOTAL, stream,
                     edge_feat, tidx, geo, W1, b1, g1, be1, W2, b2, g2, be2, W3,
                     acc, cnt, T, numTiles);

  int eBlocks = (E + 7) / 8;
  hipLaunchKernelGGL(edge_finalize_kernel, dim3(eBlocks), dim3(256), 0, stream,
                     edge_feat, acc, cnt, b3, gn, bn, (float*)d_out, E);
}
